// LoraBigNet_29068338659345
// MI455X (gfx1250) — matrix operand order repaired and checked by reading, not compiled, not run on hardware
//
#include <hip/hip_runtime.h>

// ---------------------------------------------------------------------------
// LoRA BigNet on MI455X (gfx1250): 18x fused fp16-WMMA GEMM layers
//   y = act( fp16(h) @ W^T + b + (h @ A^T) @ Bm^T [+ residual] )
// main path: v_wmma_f32_16x16x32_f16, double-buffered LDS tiles,
// W staged via GLOBAL_LOAD_ASYNC_TO_LDS (ASYNCcnt), wave32
// ---------------------------------------------------------------------------

#define N_ROWS 16384
#define DIM    1024
#define RANK   32

typedef _Float16 half_t;
typedef __attribute__((ext_vector_type(16))) _Float16 v16h;
typedef __attribute__((ext_vector_type(4)))  _Float16 v4h;
typedef __attribute__((ext_vector_type(8)))  float    v8f;
typedef __attribute__((ext_vector_type(4)))  int      v4i;

union FragH { v16h v; uint4 q[2]; };

static constexpr int BM = 128;
static constexpr int BN = 128;
static constexpr int BK = 32;
static constexpr int LSTR = 48;   // halves per LDS row (32 data + 16 pad), 96B => 16B-aligned rows

#if defined(__gfx1250__) && __has_builtin(__builtin_amdgcn_global_load_async_to_lds_b128)
#define USE_ASYNC_LDS 1
#else
#define USE_ASYNC_LDS 0
#endif

#if USE_ASYNC_LDS
#if __has_builtin(__builtin_amdgcn_s_wait_asynccnt)
#define WAIT_ASYNC() __builtin_amdgcn_s_wait_asynccnt(0)
#else
#define WAIT_ASYNC() asm volatile("s_wait_asynccnt 0" ::: "memory")
#endif
#else
#define WAIT_ASYNC() do {} while (0)
#endif

// ---------------------------------------------------------------------------
// T = X @ A^T   (X: [N,D] fp32, A: [R,D] fp32, T: [N,R] fp32)
// one wave per row, one lane per rank component
// ---------------------------------------------------------------------------
__global__ __launch_bounds__(256) void lora_t_kernel(const float* __restrict__ X,
                                                     const float* __restrict__ Al,
                                                     float* __restrict__ T)
{
    const int lane = threadIdx.x & 31;
    const int wave = threadIdx.x >> 5;
    const int row  = blockIdx.x * 8 + wave;

    const float* xr = X  + (size_t)row  * DIM;
    const float* ar = Al + (size_t)lane * DIM;

    float acc = 0.0f;
    for (int k = 0; k < DIM; k += 4) {
        float4 xv = *(const float4*)(xr + k);   // wave-uniform, broadcast
        float4 av = *(const float4*)(ar + k);
        acc += xv.x * av.x + xv.y * av.y + xv.z * av.z + xv.w * av.w;
    }
    T[(size_t)row * RANK + lane] = acc;
}

// ---------------------------------------------------------------------------
// Fused GEMM: Y[m,n] = act( sum_k fp16(X[m,k]) * W[n,k]  + b[n]
//                           + sum_r T[m,r] * Bm[n,r]  (+ res[m,n]) )
// block = 256 threads = 8 waves; block tile 128x128, K-step 32, 2x LDS buffers
// ---------------------------------------------------------------------------
__global__ __launch_bounds__(256) void gemm_lora_kernel(
    const float*  __restrict__ X,
    const half_t* __restrict__ Wl,
    const half_t* __restrict__ bl,
    const float*  __restrict__ T,
    const float*  __restrict__ BmL,
    const float*  __restrict__ res,
    float*        __restrict__ Y,
    int do_relu)
{
    __shared__ __align__(16) half_t Ash[2][BM * LSTR];  // [buf][row][k]
    __shared__ __align__(16) half_t Bsh[2][BN * LSTR];  // [buf][n][k]  (W orientation)

    const int tid      = threadIdx.x;
    const int lane     = tid & 31;
    const int wave     = tid >> 5;
    const int half_sel = lane >> 4;   // 0: lanes 0-15, 1: lanes 16-31
    const int l16      = lane & 15;

    const int n0 = blockIdx.x * BN;
    const int m0 = blockIdx.y * BM;

    // per-thread staging coordinates
    int arow[4], akc[4];
    #pragma unroll
    for (int c = 0; c < 4; ++c) {
        int cid = tid + 256 * c;            // 0..1023 chunks of 4 floats
        arow[c] = cid >> 3;
        akc[c]  = (cid & 7) << 2;
    }
    int brow[2], bkc[2];
    #pragma unroll
    for (int c = 0; c < 2; ++c) {
        int cid = tid + 256 * c;            // 0..511 chunks of 8 halves
        brow[c] = cid >> 2;
        bkc[c]  = (cid & 3) << 3;
    }

// stage W tile at k-offset K0 into Bsh[BUF] (async to LDS when available)
// builtin signature (from toolchain): (v4i AS1* src, v4i AS3* dst, imm, imm)
#if USE_ASYNC_LDS
#define STAGE_W(K0, BUF)                                                           \
    do {                                                                           \
        _Pragma("unroll")                                                          \
        for (int c = 0; c < 2; ++c) {                                              \
            __builtin_amdgcn_global_load_async_to_lds_b128(                        \
                (__attribute__((address_space(1))) v4i*)(v4i*)(uintptr_t)          \
                    (Wl + (size_t)(n0 + brow[c]) * DIM + (K0) + bkc[c]),           \
                (__attribute__((address_space(3))) v4i*)(v4i*)                     \
                    (&Bsh[BUF][brow[c] * LSTR + bkc[c]]),                          \
                0, 0);                                                             \
        }                                                                          \
    } while (0)
#else
#define STAGE_W(K0, BUF)                                                           \
    do {                                                                           \
        _Pragma("unroll")                                                          \
        for (int c = 0; c < 2; ++c) {                                              \
            *(uint4*)&Bsh[BUF][brow[c] * LSTR + bkc[c]] =                          \
                *(const uint4*)(Wl + (size_t)(n0 + brow[c]) * DIM + (K0) + bkc[c]);\
        }                                                                          \
    } while (0)
#endif

#define LOAD_X(K0, XR)                                                             \
    do {                                                                           \
        _Pragma("unroll")                                                          \
        for (int c = 0; c < 4; ++c)                                                \
            XR[c] = *(const float4*)(X + (size_t)(m0 + arow[c]) * DIM + (K0) + akc[c]); \
    } while (0)

#define STORE_X(BUF, XR)                                                           \
    do {                                                                           \
        _Pragma("unroll")                                                          \
        for (int c = 0; c < 4; ++c) {                                              \
            v4h hv = { (half_t)XR[c].x, (half_t)XR[c].y,                           \
                       (half_t)XR[c].z, (half_t)XR[c].w };                         \
            *(v4h*)&Ash[BUF][arow[c] * LSTR + akc[c]] = hv;                        \
        }                                                                          \
    } while (0)

    v8f acc[8];
    #pragma unroll
    for (int j = 0; j < 8; ++j)
        #pragma unroll
        for (int e = 0; e < 8; ++e) acc[j][e] = 0.0f;

    // --- prologue: stage tile 0 into buffer 0 ---
    float4 xr[4];
    STAGE_W(0, 0);
    LOAD_X(0, xr);
    STORE_X(0, xr);
    WAIT_ASYNC();
    __syncthreads();

    int buf = 0;
    for (int k0 = 0; k0 < DIM; k0 += BK) {
        const int  nxt      = buf ^ 1;
        const bool has_next = (k0 + BK) < DIM;

        // issue next tile's copies first (overlap with WMMA below)
        if (has_next) {
            STAGE_W(k0 + BK, nxt);   // ASYNCcnt path: no VGPR round-trip
            LOAD_X(k0 + BK, xr);     // LOADcnt path: needs f32->f16 convert
        }

        // --- A fragment (16x32, ISA 16-bit A striping) ---
        FragH a;
        const half_t* ar = &Ash[buf][(wave * 16 + l16) * LSTR + half_sel * 8];
        a.q[0] = *(const uint4*)(ar);
        a.q[1] = *(const uint4*)(ar + 16);

        // --- 8 WMMAs across the 128-wide N strip ---
        #pragma unroll
        for (int j = 0; j < 8; ++j) {
            FragH bf;
            const half_t* br = &Bsh[buf][(j * 16 + l16) * LSTR + half_sel * 8];
            bf.q[0] = *(const uint4*)(br);
            bf.q[1] = *(const uint4*)(br + 16);
            acc[j] = __builtin_amdgcn_wmma_f32_16x16x32_f16(
                false, a.v, false, bf.v, (short)0, acc[j], false, false);
        }

        if (has_next) {
            STORE_X(nxt, xr);        // convert + ds_store into next buffer
            WAIT_ASYNC();            // async W writes for nxt are in LDS
        }
        __syncthreads();             // s_wait_dscnt 0 + barrier: publishes nxt,
        buf = nxt;                   // and retires all reads of old buffer
    }

    // --- epilogue: rank-32 LoRA correction, register-blocked over T rows ---
    // C/D layout: VGPR v holds (M = v + 8*half_sel, N = l16) of each 16x16 tile
    const int rowbase = m0 + wave * 16 + half_sel * 8;
    #pragma unroll 4
    for (int r = 0; r < RANK; ++r) {
        float tv[8];
        #pragma unroll
        for (int v = 0; v < 8; ++v)
            tv[v] = T[(size_t)(rowbase + v) * RANK + r];
        #pragma unroll
        for (int j = 0; j < 8; ++j) {
            float bm = BmL[(size_t)(n0 + j * 16 + l16) * RANK + r];
            #pragma unroll
            for (int v = 0; v < 8; ++v) acc[j][v] += tv[v] * bm;
        }
    }

    // --- bias + residual + relu + store (fp32 activations) ---
    #pragma unroll
    for (int j = 0; j < 8; ++j) {
        int col = n0 + j * 16 + l16;
        float bj = (float)bl[col];
        #pragma unroll
        for (int v = 0; v < 8; ++v) {
            int row = rowbase + v;
            float val = acc[j][v] + bj;
            if (res) val += res[(size_t)row * DIM + col];
            if (do_relu) val = val > 0.0f ? val : 0.0f;
            Y[(size_t)row * DIM + col] = val;
        }
    }
#undef STAGE_W
#undef LOAD_X
#undef STORE_X
}

// ---------------------------------------------------------------------------
// In-place LayerNorm, one block (256 threads) per row of 1024
// ---------------------------------------------------------------------------
__global__ __launch_bounds__(256) void ln_kernel(float* __restrict__ H,
                                                 const float* __restrict__ g,
                                                 const float* __restrict__ beta)
{
    __shared__ float red[16];
    const int tid  = threadIdx.x;
    const int lane = tid & 31;
    const int wave = tid >> 5;
    float* hr = H + (size_t)blockIdx.x * DIM;

    float4 x = *(const float4*)(hr + tid * 4);
    float s  = x.x + x.y + x.z + x.w;
    float s2 = x.x * x.x + x.y * x.y + x.z * x.z + x.w * x.w;
    #pragma unroll
    for (int off = 16; off; off >>= 1) {
        s  += __shfl_xor(s,  off, 32);
        s2 += __shfl_xor(s2, off, 32);
    }
    if (lane == 0) { red[wave] = s; red[wave + 8] = s2; }
    __syncthreads();
    if (wave == 0) {
        float a  = (lane < 8) ? red[lane]     : 0.0f;
        float b2 = (lane < 8) ? red[lane + 8] : 0.0f;
        #pragma unroll
        for (int off = 4; off; off >>= 1) {
            a  += __shfl_xor(a,  off, 32);
            b2 += __shfl_xor(b2, off, 32);
        }
        if (lane == 0) { red[0] = a; red[1] = b2; }
    }
    __syncthreads();
    float mu  = red[0] * (1.0f / DIM);
    float var = red[1] * (1.0f / DIM) - mu * mu;
    float inv = rsqrtf(var + 1e-5f);

    float4 gg = *(const float4*)(g    + tid * 4);
    float4 bb = *(const float4*)(beta + tid * 4);
    x.x = (x.x - mu) * inv * gg.x + bb.x;
    x.y = (x.y - mu) * inv * gg.y + bb.y;
    x.z = (x.z - mu) * inv * gg.z + bb.z;
    x.w = (x.w - mu) * inv * gg.w + bb.w;
    *(float4*)(hr + tid * 4) = x;
}

// ---------------------------------------------------------------------------
extern "C" void kernel_launch(void* const* d_in, const int* in_sizes, int n_in,
                              void* d_out, int out_size, void* d_ws, size_t ws_size,
                              hipStream_t stream)
{
    (void)in_sizes; (void)n_in; (void)out_size; (void)ws_size;

    const float*  x    = (const float*) d_in[0];
    const half_t* W    = (const half_t*)d_in[1];
    const half_t* b    = (const half_t*)d_in[2];
    const float*  A    = (const float*) d_in[3];
    const float*  Bm   = (const float*) d_in[4];
    const float*  ln_g = (const float*) d_in[5];
    const float*  ln_b = (const float*) d_in[6];
    float*        out  = (float*)d_out;

    char* ws = (char*)d_ws;
    const size_t actBytes = (size_t)N_ROWS * DIM * sizeof(float);
    float* bufs[3] = { (float*)ws,
                       (float*)(ws + actBytes),
                       (float*)(ws + 2 * actBytes) };
    float* Tbuf = (float*)(ws + 3 * actBytes);   // [N, R] fp32

    dim3 bDim(256);
    dim3 gGemm(DIM / BN, N_ROWS / BM);   // (8, 128)
    dim3 gT(N_ROWS / 8);
    dim3 gLN(N_ROWS);

    const float* h = x;
    for (int blk = 0; blk < 6; ++blk) {
        // two scratch buffers not aliasing current h
        float* f[2]; int c = 0;
        for (int i = 0; i < 3 && c < 2; ++i)
            if ((const float*)bufs[i] != h) f[c++] = bufs[i];

        const int base = 3 * blk;

        // y1 = relu(lora_linear(h, base)) -> f[0]
        lora_t_kernel<<<gT, bDim, 0, stream>>>(h, A + (size_t)base * RANK * DIM, Tbuf);
        gemm_lora_kernel<<<gGemm, bDim, 0, stream>>>(
            h, W + (size_t)base * DIM * DIM, b + (size_t)base * DIM,
            Tbuf, Bm + (size_t)base * DIM * RANK, nullptr, f[0], 1);

        // y2 = relu(lora_linear(y1, base+1)) -> f[1]
        lora_t_kernel<<<gT, bDim, 0, stream>>>(f[0], A + (size_t)(base + 1) * RANK * DIM, Tbuf);
        gemm_lora_kernel<<<gGemm, bDim, 0, stream>>>(
            f[0], W + (size_t)(base + 1) * DIM * DIM, b + (size_t)(base + 1) * DIM,
            Tbuf, Bm + (size_t)(base + 1) * DIM * RANK, nullptr, f[1], 1);

        // y3 = lora_linear(y2, base+2) + h -> dst  (no relu)
        float* dst = (blk == 5) ? out : f[0];
        lora_t_kernel<<<gT, bDim, 0, stream>>>(f[1], A + (size_t)(base + 2) * RANK * DIM, Tbuf);
        gemm_lora_kernel<<<gGemm, bDim, 0, stream>>>(
            f[1], W + (size_t)(base + 2) * DIM * DIM, b + (size_t)(base + 2) * DIM,
            Tbuf, Bm + (size_t)(base + 2) * DIM * RANK, h, dst, 0);

        if (blk < 5) {
            ln_kernel<<<gLN, bDim, 0, stream>>>(dst,
                ln_g + (size_t)blk * DIM, ln_b + (size_t)blk * DIM);
            h = dst;
        }
    }
}
